// EnSSM_120259084380
// MI455X (gfx1250) — compile-verified, matrix operand order checked by reading, not run on hardware
//
#include <hip/hip_runtime.h>
#include <cstdint>

// ---------------------------------------------------------------------------
// Types for CDNA5 WMMA (wave32, 16x16x32 f16 -> f32)
// ---------------------------------------------------------------------------
typedef _Float16 f16;
typedef __attribute__((ext_vector_type(16))) _Float16 v16h;
typedef __attribute__((ext_vector_type(8)))  float    v8f;

union Frag {
    v16h     v;
    uint32_t u[8];
};

#define BM 64
#define BN 128
#define BK 64
#define TB 256   // 8 waves

// ---------------------------------------------------------------------------
// gfx1250 async global->LDS 128-bit copy (ASYNCcnt-tracked, no VGPR staging).
// Inline asm: the clang builtin declares its pointer params in the CUDA
// __device__/__shared__ language address spaces, which are not spellable from
// HIP source; asm bypasses the type check. VDST = 32-bit LDS byte address,
// VADDR = 64-bit global address (ISA 15.18.3, opcode 98).
// ---------------------------------------------------------------------------
__device__ __forceinline__ void async_b128(void* lds, const void* g) {
    const uint32_t l =
        (uint32_t)(uintptr_t)(__attribute__((address_space(3))) void*)lds;
    asm volatile("global_load_async_to_lds_b128 %0, %1, off"
                 :: "v"(l), "v"(g) : "memory");
}

template <int N>
__device__ __forceinline__ void wait_async() {
#if __has_builtin(__builtin_amdgcn_s_wait_asynccnt)
    __builtin_amdgcn_s_wait_asynccnt(N);
#else
    asm volatile("s_wait_asynccnt %0" :: "i"(N) : "memory");
#endif
}

// ---------------------------------------------------------------------------
// Fragment load from LDS (row-major [row][K] f16 stored as uint32 pairs).
// CDNA5 16-bit A-matrix layout (ISA 7.12.2): lane l -> row = l%16, half = l/16;
// packed pair v (0..3): k = 2v + half*8 (+kk), v (4..7): k = 16 + 2(v-4) + half*8.
// NT GEMM: B-fragment from W rows uses the identical addressing.
// ---------------------------------------------------------------------------
__device__ __forceinline__ void load_frag(const uint32_t (*S)[BK / 2],
                                          int rowBase, int lane, int kkW, Frag& f) {
    const int r    = rowBase + (lane & 15);
    const int base = kkW + (lane >> 4) * 4;   // uint index: kk/2 + half*4
#pragma unroll
    for (int v = 0; v < 4; ++v) f.u[v]     = S[r][base + v];
#pragma unroll
    for (int v = 0; v < 4; ++v) f.u[4 + v] = S[r][base + 8 + v];
}

// ---------------------------------------------------------------------------
// GEMM: C[M][ldc] = A[M][lda] * W[N][ldw]^T   (A,W f16 zero-padded; C f32)
// lda = ldw = Kpad (multiple of 64). M multiple of 64. N guarded on store.
// Double-buffered LDS stages filled by async-to-LDS DMA.
// ---------------------------------------------------------------------------
__global__ __launch_bounds__(TB)
void gemm_f16_wmma(const f16* __restrict__ A, int lda,
                   const f16* __restrict__ W, int ldw,
                   float* __restrict__ C, int ldc,
                   int N, int Kpad) {
    __shared__ uint32_t As[2][BM][BK / 2];   // 16 KB
    __shared__ uint32_t Bs[2][BN][BK / 2];   // 32 KB

    const int  tid  = threadIdx.x;
    const int  lane = tid & 31;
    const int  wave = tid >> 5;
    const int  wm   = wave & 1;    // 2 waves over M
    const int  wn   = wave >> 1;   // 4 waves over N
    const long m0   = (long)blockIdx.y * BM;
    const int  n0   = blockIdx.x * BN;

    // issue one K-tile's async copies: 2 A-slots + 4 B-slots of 16B per thread
    auto issue = [&](int kt, int buf) {
        const long kbase = (long)kt * BK;
#pragma unroll
        for (int i = 0; i < 2; ++i) {
            const int s = tid + i * TB, row = s >> 3, q = s & 7;
            async_b128(&As[buf][row][q * 4], A + (m0 + row) * (long)lda + kbase + q * 8);
        }
#pragma unroll
        for (int i = 0; i < 4; ++i) {
            const int s = tid + i * TB, row = s >> 3, q = s & 7;
            async_b128(&Bs[buf][row][q * 4], W + (long)(n0 + row) * ldw + kbase + q * 8);
        }
    };

    const v8f vzero = {0.f, 0.f, 0.f, 0.f, 0.f, 0.f, 0.f, 0.f};
    v8f acc[2][2];
#pragma unroll
    for (int i = 0; i < 2; ++i)
#pragma unroll
        for (int j = 0; j < 2; ++j) acc[i][j] = vzero;

    const int ktCount = Kpad / BK;
    issue(0, 0);

    for (int kt = 0; kt < ktCount; ++kt) {
        const int cb = kt & 1;
        if (kt + 1 < ktCount) {
            issue(kt + 1, cb ^ 1);   // prefetch next tile into other stage
            wait_async<6>();         // tile kt's 6 copies done; kt+1's still in flight
        } else {
            wait_async<0>();
        }
        __syncthreads();

#pragma unroll
        for (int kk = 0; kk < BK; kk += 32) {
            Frag a0, a1, b0, b1;
            load_frag(As[cb], wm * 32,      lane, kk / 2, a0);
            load_frag(As[cb], wm * 32 + 16, lane, kk / 2, a1);
            load_frag(Bs[cb], wn * 32,      lane, kk / 2, b0);
            load_frag(Bs[cb], wn * 32 + 16, lane, kk / 2, b1);
            acc[0][0] = __builtin_amdgcn_wmma_f32_16x16x32_f16(false, a0.v, false, b0.v,
                                                               (short)0, acc[0][0], false, false);
            acc[0][1] = __builtin_amdgcn_wmma_f32_16x16x32_f16(false, a0.v, false, b1.v,
                                                               (short)0, acc[0][1], false, false);
            acc[1][0] = __builtin_amdgcn_wmma_f32_16x16x32_f16(false, a1.v, false, b0.v,
                                                               (short)0, acc[1][0], false, false);
            acc[1][1] = __builtin_amdgcn_wmma_f32_16x16x32_f16(false, a1.v, false, b1.v,
                                                               (short)0, acc[1][1], false, false);
        }
        __syncthreads();   // stage free before it is overwritten next iteration
    }

    // C/D layout: lane l -> n = l%16; VGPR r -> m = r + 8*(l/16)
    const int half = lane >> 4, ln = lane & 15;
#pragma unroll
    for (int i = 0; i < 2; ++i) {
#pragma unroll
        for (int j = 0; j < 2; ++j) {
            const int nn = n0 + wn * 32 + j * 16 + ln;
            if (nn < N) {
                const long mb = m0 + wm * 32 + i * 16 + half * 8;
#pragma unroll
                for (int r = 0; r < 8; ++r)
                    C[(mb + r) * (long)ldc + nn] = acc[i][j][r];
            }
        }
    }
}

// ---------------------------------------------------------------------------
// Casts / packing kernels
// ---------------------------------------------------------------------------
__global__ void cast_act_f16(const float* __restrict__ src, int lds, int off,
                             int K, int Kpad, f16* __restrict__ dst, long total) {
    const long idx = (long)blockIdx.x * blockDim.x + threadIdx.x;
    if (idx >= total) return;
    const long m = idx / Kpad;
    const int  k = (int)(idx % Kpad);
    dst[idx] = (k < K) ? (f16)src[m * lds + off + k] : (f16)0.f;
}

__global__ void cast_w_f16(const float* __restrict__ src, int N, int K,
                           int Kpad, f16* __restrict__ dst, long total) {
    const long idx = (long)blockIdx.x * blockDim.x + threadIdx.x;
    if (idx >= total) return;
    const long n = idx / Kpad;
    const int  k = (int)(idx % Kpad);
    dst[idx] = (n < N && k < K) ? (f16)src[n * K + k] : (f16)0.f;
}

// im2col for width-3 conv, pad 1, K index = c*3 + j  (matches (d, d_prev, 3) flat)
__global__ void im2col3_f16(const float* __restrict__ x, int dprev, int L,
                            int K3, int Kpad, f16* __restrict__ dst, long total) {
    const long idx = (long)blockIdx.x * blockDim.x + threadIdx.x;
    if (idx >= total) return;
    const long m = idx / Kpad;
    const int  k = (int)(idx % Kpad);
    float v = 0.f;
    if (k < K3) {
        const int c = k / 3, j = k % 3;
        const int l = (int)(m % (long)L);
        const int ls = l + j - 1;
        if (ls >= 0 && ls < L) v = x[(m + j - 1) * dprev + c];
    }
    dst[idx] = (f16)v;
}

// ---------------------------------------------------------------------------
// Elementwise / fused kernels
// ---------------------------------------------------------------------------
__global__ void cnn_post(const float* __restrict__ h, const float* __restrict__ sc,
                         const float* __restrict__ g, const float* __restrict__ b,
                         const float* __restrict__ mn, const float* __restrict__ vr,
                         float* __restrict__ out, int d, long total) {
    const long idx = (long)blockIdx.x * blockDim.x + threadIdx.x;
    if (idx >= total) return;
    const int o = (int)(idx % d);
    float t = (h[idx] - mn[o]) * rsqrtf(vr[o] + 1e-5f) * g[o] + b[o];
    t = fmaxf(t, 0.f);
    out[idx] = t + sc[idx];
}

__global__ void convm_silu(const float* __restrict__ xz, int twoDi, int di,
                           const float* __restrict__ w, const float* __restrict__ bias,
                           float* __restrict__ u, int L, long total) {
    const long idx = (long)blockIdx.x * blockDim.x + threadIdx.x;
    if (idx >= total) return;
    const long m  = idx / di;
    const int  ch = (int)(idx % di);
    const int  l  = (int)(m % (long)L);
    float acc = bias[ch];
#pragma unroll
    for (int j = 0; j < 4; ++j) {
        const int ls = l + j - 3;
        if (ls >= 0) acc += w[ch * 4 + j] * xz[(m + j - 3) * twoDi + ch];
    }
    u[idx] = acc / (1.f + __expf(-acc));
}

// Selective scan: one thread per (batch, channel); 16-state h in registers.
// Reads dt_raw (pre-bias), u (overwritten in place with y_scan).
__global__ void scan_kernel(const float* __restrict__ dtraw, const float* __restrict__ dtb,
                            const float* __restrict__ xdb, int ldx, int dtrank,
                            const float* __restrict__ Alog, const float* __restrict__ Dp,
                            float* __restrict__ u_y, int di, int L, int B) {
    const int t = blockIdx.x * blockDim.x + threadIdx.x;
    if (t >= B * di) return;
    const int b = t / di, ch = t % di;
    float Av[16], h[16];
#pragma unroll
    for (int n = 0; n < 16; ++n) {
        Av[n] = -__expf(Alog[(long)ch * 16 + n]);
        h[n]  = 0.f;
    }
    const float bch = dtb[ch], Dch = Dp[ch];
    for (int l = 0; l < L; ++l) {
        const long  m  = (long)b * L + l;
        const float dr = dtraw[m * di + ch] + bch;
        const float dt = (dr > 20.f) ? dr : log1pf(__expf(dr));
        const float uv = u_y[m * di + ch];
        const float du = dt * uv;
        const float* row = xdb + m * ldx + dtrank;
        float y = 0.f;
#pragma unroll
        for (int n = 0; n < 16; ++n) {
            h[n] = __expf(dt * Av[n]) * h[n] + du * row[n];
            y += h[n] * row[16 + n];
        }
        u_y[m * di + ch] = y + uv * Dch;
    }
}

__global__ void gate_kernel(const float* __restrict__ yscan, const float* __restrict__ xz,
                            int twoDi, int di, float* __restrict__ out, long total) {
    const long idx = (long)blockIdx.x * blockDim.x + threadIdx.x;
    if (idx >= total) return;
    const long m  = idx / di;
    const int  ch = (int)(idx % di);
    const float z = xz[m * twoDi + di + ch];
    out[idx] = yscan[idx] * (z / (1.f + __expf(-z)));
}

__global__ void add_ln(const float* __restrict__ cnn, const float* __restrict__ mam,
                       const float* __restrict__ g, const float* __restrict__ b,
                       float* __restrict__ out, int d, long M) {
    const long m = (long)blockIdx.x * blockDim.x + threadIdx.x;
    if (m >= M) return;
    const float* row = mam + m * d;
    float s = 0.f;
    for (int o = 0; o < d; ++o) s += row[o];
    const float mean = s / d;
    float v = 0.f;
    for (int o = 0; o < d; ++o) { const float t = row[o] - mean; v += t * t; }
    const float rs = rsqrtf(v / d + 1e-6f);
    const float* crow = cnn + m * d;
    float* orow = out + m * d;
    for (int o = 0; o < d; ++o)
        orow[o] = crow[o] + (row[o] - mean) * rs * g[o] + b[o];
}

// ---------------------------------------------------------------------------
// Host driver
// ---------------------------------------------------------------------------
static inline int padto(int x, int a) { return (x + a - 1) / a * a; }
static inline unsigned nb(long total) { return (unsigned)((total + 255) / 256); }

extern "C" void kernel_launch(void* const* d_in, const int* in_sizes, int n_in,
                              void* d_out, int out_size, void* d_ws, size_t ws_size,
                              hipStream_t stream) {
    (void)in_sizes; (void)n_in; (void)out_size; (void)ws_size;
    constexpr int  Bb = 16, L = 512, NST = 16;
    constexpr long M  = (long)Bb * L;   // 8192 tokens (multiple of BM)
    const int outs[3] = {256, 512, 512};

    // ---- carve workspace ----
    char* p = (char*)d_ws;
    auto carve = [&](size_t bytes) -> void* {
        void* r = (void*)p;
        p += (bytes + 255) & ~(size_t)255;
        return r;
    };
    float* F_XA  = (float*)carve(M * 512  * sizeof(float));
    float* F_XB  = (float*)carve(M * 512  * sizeof(float));
    float* F_CNN = (float*)carve(M * 512  * sizeof(float));
    float* F_TMP = (float*)carve(M * 512  * sizeof(float));
    float* F_SC  = (float*)carve(M * 512  * sizeof(float));
    float* F_XZ  = (float*)carve(M * 2048 * sizeof(float));
    float* F_U   = (float*)carve(M * 1024 * sizeof(float));
    float* F_DT  = (float*)carve(M * 1024 * sizeof(float));
    float* F_XDB = (float*)carve(M * 64   * sizeof(float));
    float* F_MAM = (float*)carve(M * 512  * sizeof(float));
    f16*   A_BF  = (f16*)carve(M * 1536 * sizeof(f16));
    f16*   W_BF  = (f16*)carve((size_t)4 * 1024 * 1024 * sizeof(f16));

    auto gemm = [&](const f16* A, int Kpad, const f16* W, float* C, int ldc, int N) {
        dim3 grid(padto(N, BN) / BN, (unsigned)(M / BM));
        gemm_f16_wmma<<<grid, TB, 0, stream>>>(A, Kpad, W, Kpad, C, ldc, N, Kpad);
    };

    const float* cur = (const float*)d_in[0];
    int pi = 1;
    int dprev = 80;

    for (int layer = 0; layer < 3; ++layer) {
        const int  d      = outs[layer];
        const int  di     = 2 * d;              // d_inner
        const int  dtr    = (d + 15) / 16;      // dt_rank
        const bool has_sc = (dprev != d);

        const float* P[17];
        const int np = has_sc ? 17 : 16;
        for (int i = 0; i < np; ++i) P[i] = (const float*)d_in[pi + i];
        pi += np;
        int q = 0;
        const float* conv1_w    = P[q++];
        const float* sc_w       = has_sc ? P[q++] : nullptr;
        const float* bn_g       = P[q++];
        const float* bn_b       = P[q++];
        const float* bn_m       = P[q++];
        const float* bn_v       = P[q++];
        const float* in_proj_w  = P[q++];
        const float* convm_w    = P[q++];
        const float* convm_b    = P[q++];
        const float* x_proj_w   = P[q++];
        const float* dt_proj_w  = P[q++];
        const float* dt_proj_b  = P[q++];
        const float* A_log      = P[q++];
        const float* Dp         = P[q++];
        const float* out_proj_w = P[q++];
        const float* ln_g       = P[q++];
        const float* ln_b       = P[q++];

        // ---- CNN: conv1 (im2col GEMM) + BN + ReLU + shortcut ----
        const int K3 = 3 * dprev, K3p = padto(K3, BK);
        im2col3_f16<<<nb(M * K3p), 256, 0, stream>>>(cur, dprev, L, K3, K3p, A_BF, M * K3p);
        {
            const long wt = (long)padto(d, BN) * K3p;
            cast_w_f16<<<nb(wt), 256, 0, stream>>>(conv1_w, d, K3, K3p, W_BF, wt);
        }
        gemm(A_BF, K3p, W_BF, F_TMP, d, d);

        const float* scptr;
        if (has_sc) {
            const int Kp = padto(dprev, BK);
            cast_act_f16<<<nb(M * Kp), 256, 0, stream>>>(cur, dprev, 0, dprev, Kp, A_BF, M * Kp);
            const long wt = (long)padto(d, BN) * Kp;
            cast_w_f16<<<nb(wt), 256, 0, stream>>>(sc_w, d, dprev, Kp, W_BF, wt);
            gemm(A_BF, Kp, W_BF, F_SC, d, d);
            scptr = F_SC;
        } else {
            scptr = cur;   // identity shortcut (dprev == d)
        }
        cnn_post<<<nb(M * d), 256, 0, stream>>>(F_TMP, scptr, bn_g, bn_b, bn_m, bn_v,
                                                F_CNN, d, M * d);

        // ---- in_proj: xz = cnn @ in_proj_w^T  (N = 2*d_inner) ----
        const int dp = padto(d, BK);
        cast_act_f16<<<nb(M * dp), 256, 0, stream>>>(F_CNN, d, 0, d, dp, A_BF, M * dp);
        {
            const long wt = (long)padto(2 * di, BN) * dp;
            cast_w_f16<<<nb(wt), 256, 0, stream>>>(in_proj_w, 2 * di, d, dp, W_BF, wt);
        }
        gemm(A_BF, dp, W_BF, F_XZ, 2 * di, 2 * di);

        // ---- depthwise causal conv (width 4) + SiLU -> u ----
        convm_silu<<<nb(M * di), 256, 0, stream>>>(F_XZ, 2 * di, di, convm_w, convm_b,
                                                   F_U, L, M * di);

        // ---- x_proj: xdb = u @ x_proj_w^T  (N = dt_rank + 32) ----
        const int dip = padto(di, BK);
        const int nx  = dtr + 2 * NST;
        cast_act_f16<<<nb(M * dip), 256, 0, stream>>>(F_U, di, 0, di, dip, A_BF, M * dip);
        {
            const long wt = (long)padto(nx, BN) * dip;
            cast_w_f16<<<nb(wt), 256, 0, stream>>>(x_proj_w, nx, di, dip, W_BF, wt);
        }
        gemm(A_BF, dip, W_BF, F_XDB, nx, nx);

        // ---- dt_proj: dt_raw = dtr @ dt_proj_w^T ----
        const int krp = padto(dtr, BK);
        cast_act_f16<<<nb(M * krp), 256, 0, stream>>>(F_XDB, nx, 0, dtr, krp, A_BF, M * krp);
        {
            const long wt = (long)padto(di, BN) * krp;
            cast_w_f16<<<nb(wt), 256, 0, stream>>>(dt_proj_w, di, dtr, krp, W_BF, wt);
        }
        gemm(A_BF, krp, W_BF, F_DT, di, di);

        // ---- selective scan (u -> y_scan in place) ----
        scan_kernel<<<nb((long)Bb * di), 256, 0, stream>>>(F_DT, dt_proj_b, F_XDB, nx, dtr,
                                                           A_log, Dp, F_U, di, L, Bb);

        // ---- gate: y = y_scan * silu(z) -> F_DT ----
        gate_kernel<<<nb(M * di), 256, 0, stream>>>(F_U, F_XZ, 2 * di, di, F_DT, M * di);

        // ---- out_proj: mam = y @ out_proj_w^T ----
        cast_act_f16<<<nb(M * dip), 256, 0, stream>>>(F_DT, di, 0, di, dip, A_BF, M * dip);
        {
            const long wt = (long)padto(d, BN) * dip;
            cast_w_f16<<<nb(wt), 256, 0, stream>>>(out_proj_w, d, di, dip, W_BF, wt);
        }
        gemm(A_BF, dip, W_BF, F_MAM, d, d);

        // ---- residual + layernorm ----
        float* dst = (layer == 2) ? (float*)d_out : ((layer & 1) ? F_XB : F_XA);
        add_ln<<<nb(M), 256, 0, stream>>>(F_CNN, F_MAM, ln_g, ln_b, dst, d, M);

        cur = dst;
        dprev = d;
    }
}
